// FastNeuralMemory_79740362817556
// MI455X (gfx1250) — compile-verified, matrix operand order checked by reading, not run on hardware
//
#include <hip/hip_runtime.h>
#include <hip/hip_bf16.h>

#define D_DIM 1024
#define B_DIM 4
#define L_DIM 2048
#define NCHUNK 32
#define CHUNK_ 64
#define NTOK (B_DIM * L_DIM)                 // 8192 token rows
#define BLD ((size_t)NTOK * D_DIM)           // 8,388,608 elements
#define BDD ((size_t)B_DIM * D_DIM * D_DIM)  // 4,194,304 elements
#define DD  ((size_t)D_DIM * D_DIM)          // 1,048,576 elements

typedef __attribute__((ext_vector_type(16))) __bf16 bf16x16;
typedef __attribute__((ext_vector_type(8)))  __bf16 bf16x8;
typedef __attribute__((ext_vector_type(4)))  __bf16 bf16x4;
typedef __attribute__((ext_vector_type(8)))  float  f32x8;
typedef __attribute__((ext_vector_type(4)))  unsigned int v4u;
typedef __attribute__((ext_vector_type(8)))  int v8i;
typedef __attribute__((ext_vector_type(4)))  int v4i;

#if defined(__gfx1250__) && __has_builtin(__builtin_amdgcn_tensor_load_to_lds)
#define HAS_TDM 1
#pragma message("CDNA5: tensor_load_to_lds builtin AVAILABLE (TDM path enabled)")
#else
#define HAS_TDM 0
#if defined(__gfx1250__)
#pragma message("CDNA5: tensor_load_to_lds builtin MISSING (fallback staging)")
#endif
#endif

#if HAS_TDM
#define TDM_WAIT(n) __builtin_amdgcn_s_wait_tensorcnt(n)
#else
#define TDM_WAIT(n)
#endif

__device__ inline float fast_rcp(float x) {
#if defined(__gfx1250__)
  return __builtin_amdgcn_rcpf(x);
#else
  return 1.f / x;
#endif
}
__device__ inline float sigmoidf_(float t) { return fast_rcp(1.f + __expf(-t)); }

__device__ inline f32x8 zero8() {
  f32x8 z;
#pragma unroll
  for (int i = 0; i < 8; ++i) z[i] = 0.f;
  return z;
}

// generic pointer to a __shared__ object: low 32 bits are the LDS byte offset
__device__ inline unsigned lds_offset_of(const void* p) {
  return (unsigned)(uintptr_t)p;
}

// ---------------------------------------------------------------------------
// Tensor Data Mover: load a 2D bf16 tile [tile_y rows x tile_x elems] from
// global (row stride = stride_elems) into LDS at lds_off, inserting LDS row
// padding via the D# pad fields. D# per CDNA5 ISA 8.3-8.5; TENSORcnt-tracked.
// ---------------------------------------------------------------------------
__device__ inline void tdm_load_2d_bf16(unsigned lds_off, const void* gptr,
                                        unsigned tile_x, unsigned tile_y,
                                        unsigned stride_elems,
                                        unsigned pad_int_code, unsigned pad_amt_code) {
#if HAS_TDM
  unsigned long long ga = (unsigned long long)(uintptr_t)gptr;
  v4u g0 = { 1u,                                     // count=1 (valid, user mode)
             lds_off,                                // lds_addr (bytes)
             (unsigned)(ga & 0xFFFFFFFFu),           // global_addr[31:0]
             (unsigned)((ga >> 32) & 0x01FFFFFFu) | (2u << 30) };  // addr hi | type=2
  const unsigned td0 = 1u << 20, td1 = 1u << 20;     // huge tensor dims: no OOB clip
  unsigned dw0 = (1u << 16)                          // data_size = 1 -> 2 bytes
               | (1u << 20)                          // pad_enable
               | (pad_int_code << 22) | (pad_amt_code << 25);
  v8i g1 = { (int)dw0,
             (int)((td0 & 0xFFFFu) << 16),                              // dim0 lo16
             (int)(((td0 >> 16) & 0xFFFFu) | ((td1 & 0xFFFFu) << 16)),  // dim0 hi / dim1 lo
             (int)(((td1 >> 16) & 0xFFFFu) | (tile_x << 16)),           // dim1 hi / tile_dim0
             (int)tile_y,                                               // tile_dim1
             (int)stride_elems,                                         // dim0_stride lo32
             0, 0 };
  v4i gz = { 0, 0, 0, 0 };
#if __clang_major__ >= 23
  v8i gz8 = { 0, 0, 0, 0, 0, 0, 0, 0 };
  __builtin_amdgcn_tensor_load_to_lds(g0, g1, gz, gz, gz8, 0);
#else
  __builtin_amdgcn_tensor_load_to_lds(g0, g1, gz, gz, 0);
#endif
#else
  (void)lds_off; (void)gptr; (void)tile_x; (void)tile_y;
  (void)stride_elems; (void)pad_int_code; (void)pad_amt_code;
#endif
}

// A-fragment (or B-fragment when tile is stored [n][k]):
// lane holds row (row0 + lane&15); K elems at kb=(lane>>4)*8: {kb..kb+7, kb+16..kb+23}
__device__ inline bf16x16 frag_row(const __bf16* base, int lane, int row0, int ld, int k0) {
  const __bf16* p = base + (row0 + (lane & 15)) * ld + k0 + ((lane >> 4) << 3);
  bf16x16 f;
#pragma unroll
  for (int j = 0; j < 8; ++j) { f[j] = p[j]; f[j + 8] = p[j + 16]; }
  return f;
}

__device__ inline f32x8 wmma_bf16(bf16x16 a, bf16x16 b, f32x8 c) {
  return __builtin_amdgcn_wmma_f32_16x16x32_bf16(false, a, false, b, (short)0, c, false, false);
}

// ---------------------------------------------------------------------------
// f32 -> bf16 bulk convert (n divisible by 4). grid = n/4/256
// ---------------------------------------------------------------------------
__global__ __launch_bounds__(256) void cvt_bf16_kernel(const float* __restrict__ src,
                                                       __bf16* __restrict__ dst, int n4) {
  int i = blockIdx.x * 256 + threadIdx.x;
  if (i < n4) {
    float4 v = ((const float4*)src)[i];
    bf16x4 o = { (__bf16)v.x, (__bf16)v.y, (__bf16)v.z, (__bf16)v.w };
    ((bf16x4*)dst)[i] = o;
  }
}

// ---------------------------------------------------------------------------
// Projection: Y = silu(Xb @ Wb^T), bf16 inputs, f32 out.
// Block tile 64 x 128, 8 waves; TDM double-buffered K-stream (DMA overlaps WMMA).
// grid = (NTOK/64) * (D/128) = 1024
// ---------------------------------------------------------------------------
__global__ __launch_bounds__(256) void proj_silu_kernel(const __bf16* __restrict__ Xb,
                                                        const __bf16* __restrict__ Wb,
                                                        float* __restrict__ Y) {
  const int XSZ = 64 * 40, WSZ = 128 * 40;
  __shared__ __align__(16) __bf16 Xs[2 * 64 * 40];
  __shared__ __align__(16) __bf16 Ws[2 * 128 * 40];
  const int tid = threadIdx.x;
  const int lane = tid & 31;
  const int wv = tid >> 5;
  const int mw = wv >> 1;
  const int nw = wv & 1;
  const int rowBase = (blockIdx.x >> 3) * 64;
  const int colBase = (blockIdx.x & 7) * 128;
  const __bf16* Xg = Xb + (size_t)rowBase * D_DIM;
  const __bf16* Wg = Wb + (size_t)colBase * D_DIM;

  f32x8 acc0 = zero8(), acc1 = zero8(), acc2 = zero8(), acc3 = zero8();

#if HAS_TDM
  if (wv == 0) {   // prologue: DMA tile 0 into buffer 0
    tdm_load_2d_bf16(lds_offset_of(Xs), Xg, 32, 64, D_DIM, 3, 3);
    tdm_load_2d_bf16(lds_offset_of(Ws), Wg, 32, 128, D_DIM, 3, 3);
  }
#endif
  for (int it = 0; it < 32; ++it) {
    const int p = it & 1;
    const int k0 = it * 32;
#if HAS_TDM
    if (wv == 0) {
      if (it < 31) {  // DMA next tile into the other buffer, then wait for current
        tdm_load_2d_bf16(lds_offset_of(Xs + (p ^ 1) * XSZ), Xg + k0 + 32, 32, 64, D_DIM, 3, 3);
        tdm_load_2d_bf16(lds_offset_of(Ws + (p ^ 1) * WSZ), Wg + k0 + 32, 32, 128, D_DIM, 3, 3);
        TDM_WAIT(2);
      } else {
        TDM_WAIT(0);
      }
    }
#else
#pragma unroll
    for (int j = 0; j < 8; ++j) {
      int idx = tid * 8 + j;
      int r = idx >> 5, c = idx & 31;
      Xs[p * XSZ + r * 40 + c] = Xg[(size_t)r * D_DIM + k0 + c];
    }
#pragma unroll
    for (int j = 0; j < 16; ++j) {
      int idx = tid * 16 + j;
      int r = idx >> 5, c = idx & 31;
      Ws[p * WSZ + r * 40 + c] = Wg[(size_t)r * D_DIM + k0 + c];
    }
#endif
    __syncthreads();
    const __bf16* Xp = Xs + p * XSZ;
    const __bf16* Wp = Ws + p * WSZ;
    bf16x16 a  = frag_row(Xp, lane, mw * 16, 40, 0);
    bf16x16 b0 = frag_row(Wp, lane, nw * 64 +  0, 40, 0);
    bf16x16 b1 = frag_row(Wp, lane, nw * 64 + 16, 40, 0);
    bf16x16 b2 = frag_row(Wp, lane, nw * 64 + 32, 40, 0);
    bf16x16 b3 = frag_row(Wp, lane, nw * 64 + 48, 40, 0);
    acc0 = wmma_bf16(a, b0, acc0);
    acc1 = wmma_bf16(a, b1, acc1);
    acc2 = wmma_bf16(a, b2, acc2);
    acc3 = wmma_bf16(a, b3, acc3);
    __syncthreads();   // WAR: everyone done with buffer p before it is re-DMAed
  }

  const int m0 = mw * 16 + ((lane >> 4) << 3);
  const int nb = colBase + nw * 64 + (lane & 15);
#pragma unroll
  for (int r = 0; r < 8; ++r) {
    size_t ro = (size_t)(rowBase + m0 + r) * D_DIM;
    float y0 = acc0[r]; Y[ro + nb +  0] = y0 * sigmoidf_(y0);
    float y1 = acc1[r]; Y[ro + nb + 16] = y1 * sigmoidf_(y1);
    float y2 = acc2[r]; Y[ro + nb + 32] = y2 * sigmoidf_(y2);
    float y3 = acc3[r]; Y[ro + nb + 48] = y3 * sigmoidf_(y3);
  }
}

// ---------------------------------------------------------------------------
// In-place row l2-normalize (eps=1e-12) + bf16 shadow copy.  grid = NTOK
// ---------------------------------------------------------------------------
__global__ __launch_bounds__(256) void l2norm_bf_kernel(float* __restrict__ Y,
                                                        __bf16* __restrict__ Ybf) {
  __shared__ float red[256];
  float* p = Y + (size_t)blockIdx.x * D_DIM;
  __bf16* pb = Ybf + (size_t)blockIdx.x * D_DIM;
  float ss = 0.f;
  for (int i = threadIdx.x; i < D_DIM; i += 256) { float v = p[i]; ss += v * v; }
  red[threadIdx.x] = ss;
  __syncthreads();
  for (int o = 128; o > 0; o >>= 1) {
    if (threadIdx.x < o) red[threadIdx.x] += red[threadIdx.x + o];
    __syncthreads();
  }
  const float inv = fast_rcp(fmaxf(sqrtf(red[0]), 1e-12f));
  for (int i = threadIdx.x; i < D_DIM; i += 256) {
    float nv = p[i] * inv;
    p[i] = nv;
    pb[i] = (__bf16)nv;
  }
}

// ---------------------------------------------------------------------------
// Transpose normalized K (f32 [b*L, D]) into KT bf16 [B][D][L].
// grid = (NTOK/64) * (D/64) = 2048; LDS tile transpose, coalesced both sides.
// ---------------------------------------------------------------------------
__global__ __launch_bounds__(256) void transpose_bf16_kernel(const float* __restrict__ src,
                                                             __bf16* __restrict__ KT) {
  __shared__ __bf16 t[64 * 72];
  const int rt = blockIdx.x >> 4;    // token tile
  const int ct = blockIdx.x & 15;    // d tile
  const int row0 = rt * 64, col0 = ct * 64;
#pragma unroll
  for (int j = 0; j < 16; ++j) {
    int idx = threadIdx.x * 16 + j;
    int r = idx >> 6, c = idx & 63;  // r = token-local, c = d-local
    t[c * 72 + r] = (__bf16)src[(size_t)(row0 + r) * D_DIM + col0 + c];
  }
  __syncthreads();
  const int b = row0 / L_DIM;
  const int l0 = row0 % L_DIM;
#pragma unroll
  for (int j = 0; j < 16; ++j) {
    int idx = threadIdx.x * 16 + j;
    int r = idx >> 6, c = idx & 63;  // r = d-local, c = token-local
    KT[(size_t)b * D_DIM * L_DIM + (size_t)(col0 + r) * L_DIM + l0 + c] = t[r * 72 + c];
  }
}

// ---------------------------------------------------------------------------
// Gates.  grid = B*NCHUNK = 128.  G: [0..127]=theta, [128..255]=eta, [256..383]=alpha
// ---------------------------------------------------------------------------
__global__ __launch_bounds__(256) void gates_kernel(const float* __restrict__ X,
    const float* __restrict__ tw, const float* __restrict__ tb,
    const float* __restrict__ ew, const float* __restrict__ eb,
    const float* __restrict__ aw, const float* __restrict__ ab,
    float* __restrict__ G) {
  __shared__ float p0[256], p1[256], p2[256];
  __shared__ float sg0[64], sg1[64], sg2[64];
  const int b = blockIdx.x >> 5;
  const int c = blockIdx.x & 31;
  const int tokenBase = b * L_DIM + c * CHUNK_;
  const int tid = threadIdx.x;
  const int tok = tid >> 2;
  const int qp = tid & 3;
  const float* xr = X + (size_t)(tokenBase + tok) * D_DIM + qp * 256;
  float s0 = 0.f, s1 = 0.f, s2 = 0.f;
  for (int i = 0; i < 256; ++i) {
    float xv = xr[i];
    s0 += xv * tw[qp * 256 + i];
    s1 += xv * ew[qp * 256 + i];
    s2 += xv * aw[qp * 256 + i];
  }
  p0[tid] = s0; p1[tid] = s1; p2[tid] = s2;
  __syncthreads();
  if (tid < 64) {
    int base = tid * 4;
    float d0 = p0[base] + p0[base + 1] + p0[base + 2] + p0[base + 3];
    float d1 = p1[base] + p1[base + 1] + p1[base + 2] + p1[base + 3];
    float d2 = p2[base] + p2[base + 1] + p2[base + 2] + p2[base + 3];
    sg0[tid] = sigmoidf_(d0 + tb[0]);
    sg1[tid] = sigmoidf_(d1 + eb[0]);
    sg2[tid] = sigmoidf_(d2 + ab[0]);
  }
  __syncthreads();
  if (tid == 0) {
    float a0 = 0.f, a1 = 0.f, a2 = 0.f;
    for (int i = 0; i < 64; ++i) { a0 += sg0[i]; a1 += sg1[i]; a2 += sg2[i]; }
    G[b * NCHUNK + c]       = a0 * (1.f / 64.f);
    G[128 + b * NCHUNK + c] = a1 * (1.f / 64.f);
    G[256 + b * NCHUNK + c] = a2 * (1.f / 64.f);
  }
}

// ---------------------------------------------------------------------------
// One scan step.  grid = B*32; each block owns 32 e-rows of W/M for one batch.
// Phase 1: pred = k @ W^T, out = q @ W^T (WMMA; TDM double-buffered tiles).
// Phase 2: gW = err^T @ k / 64 (WMMA; k^T tiles via TDM from KT), fused update.
// chunk==0: W=M=0 (no loads).
// ---------------------------------------------------------------------------
__global__ __launch_bounds__(256) void scan_step_kernel(
    const __bf16* __restrict__ Kb, const __bf16* __restrict__ Qb,
    const __bf16* __restrict__ KT, const float* __restrict__ V,
    float* __restrict__ Wst, float* __restrict__ Mst, __bf16* __restrict__ Wbf,
    const float* __restrict__ G, float* __restrict__ OB, int chunk) {
  const int KSZ = 64 * 40, WSMSZ = 32 * 40, TSZ = 64 * 72;
  __shared__ __align__(16) __bf16 Ks[2 * 64 * 40];
  __shared__ __align__(16) __bf16 Qs[2 * 64 * 40];
  __shared__ __align__(16) __bf16 Wsm[2 * 32 * 40];
  __shared__ __align__(16) __bf16 errT[32 * 72];     // [e][c]
  __shared__ __align__(16) __bf16 ks2T[2 * 64 * 72]; // [d][c] double-buffered
  const int tid = threadIdx.x;
  const int lane = tid & 31;
  const int wv = tid >> 5;
  const int b = blockIdx.x >> 5;
  const int e0 = (blockIdx.x & 31) * 32;
  const int tokenBase = b * L_DIM + chunk * CHUNK_;
  const bool first = (chunk == 0);
  float* Wb = Wst + (size_t)b * DD;
  float* Mb = Mst + (size_t)b * DD;
  __bf16* Wbfb = Wbf + (size_t)b * DD;
  const __bf16* KTb = KT + (size_t)b * D_DIM * L_DIM;
  const __bf16* Kg = Kb + (size_t)tokenBase * D_DIM;
  const __bf16* Qg = Qb + (size_t)tokenBase * D_DIM;
  const __bf16* Wg = Wbfb + (size_t)e0 * D_DIM;

  // ---- Phase 1: out tile 64(c) x 32(e) ----
  const int mw = wv & 3;   // token 16-tile
  const int nt = wv >> 2;  // e 16-tile (0..1)
  f32x8 accP = zero8(), accO = zero8();
  if (!first) {
#if HAS_TDM
    if (wv == 0) {   // prologue: tile 0 into buffer 0
      tdm_load_2d_bf16(lds_offset_of(Ks),  Kg, 32, 64, D_DIM, 3, 3);
      tdm_load_2d_bf16(lds_offset_of(Qs),  Qg, 32, 64, D_DIM, 3, 3);
      tdm_load_2d_bf16(lds_offset_of(Wsm), Wg, 32, 32, D_DIM, 3, 3);
    }
#endif
    for (int it = 0; it < 32; ++it) {
      const int p = it & 1;
      const int k0 = it * 32;
#if HAS_TDM
      if (wv == 0) {
        if (it < 31) {
          tdm_load_2d_bf16(lds_offset_of(Ks + (p ^ 1) * KSZ),  Kg + k0 + 32, 32, 64, D_DIM, 3, 3);
          tdm_load_2d_bf16(lds_offset_of(Qs + (p ^ 1) * KSZ),  Qg + k0 + 32, 32, 64, D_DIM, 3, 3);
          tdm_load_2d_bf16(lds_offset_of(Wsm + (p ^ 1) * WSMSZ), Wg + k0 + 32, 32, 32, D_DIM, 3, 3);
          TDM_WAIT(3);
        } else {
          TDM_WAIT(0);
        }
      }
#else
#pragma unroll
      for (int j = 0; j < 8; ++j) {
        int idx = tid * 8 + j;
        int r = idx >> 5, cc = idx & 31;
        Ks[p * KSZ + r * 40 + cc] = Kg[(size_t)r * D_DIM + k0 + cc];
        Qs[p * KSZ + r * 40 + cc] = Qg[(size_t)r * D_DIM + k0 + cc];
      }
#pragma unroll
      for (int j = 0; j < 4; ++j) {
        int idx = tid * 4 + j;
        int r = idx >> 5, cc = idx & 31;
        Wsm[p * WSMSZ + r * 40 + cc] = Wg[(size_t)r * D_DIM + k0 + cc];
      }
#endif
      __syncthreads();
      bf16x16 ak = frag_row(Ks + p * KSZ, lane, mw * 16, 40, 0);
      bf16x16 aq = frag_row(Qs + p * KSZ, lane, mw * 16, 40, 0);
      bf16x16 bw = frag_row(Wsm + p * WSMSZ, lane, nt * 16, 40, 0);
      accP = wmma_bf16(ak, bw, accP);
      accO = wmma_bf16(aq, bw, accO);
      __syncthreads();
    }
  }
#if HAS_TDM
  if (wv == 0) {   // phase-2 prologue DMA overlaps the epilogue below
    tdm_load_2d_bf16(lds_offset_of(ks2T), KTb + chunk * CHUNK_, 64, 64, L_DIM, 4, 3);
  }
#endif
  {
    const int m0 = mw * 16 + ((lane >> 4) << 3);
    const int n0 = nt * 16 + (lane & 15);
    bf16x8 pe;
#pragma unroll
    for (int r = 0; r < 8; ++r) {
      size_t o = (size_t)(tokenBase + m0 + r) * D_DIM + e0 + n0;
      OB[o] = accO[r];
      pe[r] = (__bf16)(accP[r] - V[o]);
    }
    *(bf16x8*)(&errT[n0 * 72 + m0]) = pe;   // one b128 store per lane
  }
  __syncthreads();

  // ---- Phase 2: gW tile 32(e) x 64(d) per d-step ----
  const float th  = G[b * NCHUNK + chunk];
  const float eta = G[128 + b * NCHUNK + chunk];
  const float al  = G[256 + b * NCHUNK + chunk];
  const float ths = th * (1.f / 64.f);
  const float ooa = 1.f - al;
  const int ewt = wv & 1;   // e 16-tile
  const int dt  = wv >> 1;  // d 16-tile (0..3)

  for (int it = 0; it < 16; ++it) {
    const int p = it & 1;
    const int d0 = it * 64;
#if HAS_TDM
    if (wv == 0) {
      if (it < 15) {
        tdm_load_2d_bf16(lds_offset_of(ks2T + (p ^ 1) * TSZ),
                         KTb + (size_t)(d0 + 64) * L_DIM + chunk * CHUNK_, 64, 64, L_DIM, 4, 3);
        TDM_WAIT(1);
      } else {
        TDM_WAIT(0);
      }
    }
#else
#pragma unroll
    for (int j = 0; j < 16; ++j) {
      int idx = tid * 16 + j;
      int rd = idx >> 6, cc = idx & 63;
      ks2T[p * TSZ + rd * 72 + cc] = KTb[(size_t)(d0 + rd) * L_DIM + chunk * CHUNK_ + cc];
    }
#endif
    __syncthreads();
    f32x8 g0 = zero8();
#pragma unroll
    for (int cs = 0; cs < 64; cs += 32) {
      bf16x16 aT = frag_row(errT, lane, ewt * 16, 72, cs);
      bf16x16 bk = frag_row(ks2T + p * TSZ, lane, dt * 16, 72, cs);
      g0 = wmma_bf16(aT, bk, g0);
    }
    const int erow0 = e0 + ewt * 16 + ((lane >> 4) << 3);
    const int dc = d0 + dt * 16 + (lane & 15);
#pragma unroll
    for (int r = 0; r < 8; ++r) {
      size_t i0 = (size_t)(erow0 + r) * D_DIM + dc;
      float m0v = first ? 0.f : Mb[i0];
      float w0v = first ? 0.f : Wb[i0];
      float mn0 = eta * m0v - ths * g0[r];
      float wn0 = ooa * w0v + mn0;
      Mb[i0] = mn0;
      Wb[i0] = wn0;
      Wbfb[i0] = (__bf16)wn0;
    }
    __syncthreads();
  }
}

// ---------------------------------------------------------------------------
// Final LayerNorm (biased var, eps=1e-5).  grid = NTOK
// ---------------------------------------------------------------------------
__global__ __launch_bounds__(256) void layernorm_kernel(const float* __restrict__ OB,
    const float* __restrict__ g, const float* __restrict__ bta, float* __restrict__ out) {
  __shared__ float r1[256], r2[256];
  const float* p = OB + (size_t)blockIdx.x * D_DIM;
  float a = 0.f, sq = 0.f;
  for (int i = threadIdx.x; i < D_DIM; i += 256) { float v = p[i]; a += v; sq += v * v; }
  r1[threadIdx.x] = a; r2[threadIdx.x] = sq;
  __syncthreads();
  for (int o = 128; o > 0; o >>= 1) {
    if (threadIdx.x < o) { r1[threadIdx.x] += r1[threadIdx.x + o]; r2[threadIdx.x] += r2[threadIdx.x + o]; }
    __syncthreads();
  }
  const float mu = r1[0] * (1.f / D_DIM);
  const float var = r2[0] * (1.f / D_DIM) - mu * mu;
  const float inv = rsqrtf(var + 1e-5f);
  for (int i = threadIdx.x; i < D_DIM; i += 256)
    out[(size_t)blockIdx.x * D_DIM + i] = (p[i] - mu) * inv * g[i] + bta[i];
}

// ---------------------------------------------------------------------------
extern "C" void kernel_launch(void* const* d_in, const int* in_sizes, int n_in,
                              void* d_out, int out_size, void* d_ws, size_t ws_size,
                              hipStream_t stream) {
  const float* x   = (const float*)d_in[0];
  const float* Wk  = (const float*)d_in[1];
  const float* Wv  = (const float*)d_in[2];
  const float* Wq  = (const float*)d_in[3];
  const float* tw  = (const float*)d_in[4];
  const float* tb  = (const float*)d_in[5];
  const float* ew  = (const float*)d_in[6];
  const float* eb  = (const float*)d_in[7];
  const float* aw  = (const float*)d_in[8];
  const float* ab  = (const float*)d_in[9];
  const float* lng = (const float*)d_in[10];
  const float* lnb = (const float*)d_in[11];
  float* out = (float*)d_out;

  // workspace (f32): K | V | Q | outbuf | Wstate | Mstate | gates
  // then (bf16): Kbf | Qbf | Wbf | xbf | KT | wkbf | wvbf | wqbf
  float* ws   = (float*)d_ws;
  float* Kbuf = ws;
  float* Vbuf = ws + BLD;
  float* Qbuf = ws + 2 * BLD;
  float* OB   = ws + 3 * BLD;
  float* Wst  = ws + 4 * BLD;
  float* Mst  = Wst + BDD;
  float* G    = Mst + BDD;            // 384 floats
  __bf16* Kbf  = (__bf16*)(G + 384);
  __bf16* Qbf  = Kbf + BLD;
  __bf16* Wbf  = Qbf + BLD;           // BDD elems
  __bf16* xbf  = Wbf + BDD;           // BLD elems
  __bf16* KT   = xbf + BLD;           // BLD elems
  __bf16* wkbf = KT + BLD;            // DD elems
  __bf16* wvbf = wkbf + DD;
  __bf16* wqbf = wvbf + DD;

  dim3 blk(256);
  cvt_bf16_kernel<<<dim3((unsigned)(BLD / 4 / 256)), blk, 0, stream>>>(x, xbf, (int)(BLD / 4));
  cvt_bf16_kernel<<<dim3((unsigned)(DD / 4 / 256)), blk, 0, stream>>>(Wk, wkbf, (int)(DD / 4));
  cvt_bf16_kernel<<<dim3((unsigned)(DD / 4 / 256)), blk, 0, stream>>>(Wv, wvbf, (int)(DD / 4));
  cvt_bf16_kernel<<<dim3((unsigned)(DD / 4 / 256)), blk, 0, stream>>>(Wq, wqbf, (int)(DD / 4));
  proj_silu_kernel<<<dim3(128 * 8), blk, 0, stream>>>(xbf, wkbf, Kbuf);
  proj_silu_kernel<<<dim3(128 * 8), blk, 0, stream>>>(xbf, wvbf, Vbuf);
  proj_silu_kernel<<<dim3(128 * 8), blk, 0, stream>>>(xbf, wqbf, Qbuf);
  l2norm_bf_kernel<<<dim3(NTOK), blk, 0, stream>>>(Kbuf, Kbf);
  l2norm_bf_kernel<<<dim3(NTOK), blk, 0, stream>>>(Qbuf, Qbf);
  transpose_bf16_kernel<<<dim3(128 * 16), blk, 0, stream>>>(Kbuf, KT);
  gates_kernel<<<dim3(B_DIM * NCHUNK), blk, 0, stream>>>(x, tw, tb, ew, eb, aw, ab, G);
  for (int c = 0; c < NCHUNK; ++c)
    scan_step_kernel<<<dim3(B_DIM * 32), blk, 0, stream>>>(Kbf, Qbf, KT, Vbuf, Wst, Mst, Wbf, G, OB, c);
  layernorm_kernel<<<dim3(NTOK), blk, 0, stream>>>(OB, lng, lnb, out);
}
